// WeightedTP_7782480740617
// MI455X (gfx1250) — matrix operand order repaired and checked by reading, not compiled, not run on hardware
//
#include <hip/hip_runtime.h>

typedef __attribute__((ext_vector_type(16))) __bf16 bf16x16;
typedef __attribute__((ext_vector_type(8)))  float  f32x8;

#define NROWS      2048
#define BLK_ROWS   16
#define NBLOCKS    (NROWS / BLK_ROWS)      // 128
#define XD         64
#define KTOT       33280                   // 64*256*2 + 512 bias rows
#define NCHUNKS    (KTOT / 32)             // 1040
#define CHUNKS_PER_WAVE (NCHUNKS / 4)      // 260  (4 K-quarters)
#define FFSTRIDE   260                     // 16B-aligned rows; 260%64=4 -> bank-spread
#define PATH_COEFF 0.04419417382415922f    // 1/sqrt(512)
#define CG110      0.5773502691896258f     // 1/sqrt(3)

static __device__ __forceinline__ unsigned short f2bf_rne(float f) {
    union { float f; unsigned int u; } x; x.f = f;
    unsigned int u = x.u;
    unsigned int r = u + 0x7FFFu + ((u >> 16) & 1u);
    if ((u & 0x7F800000u) == 0x7F800000u) r = u;   // inf/nan passthrough
    return (unsigned short)(r >> 16);
}

// ---------------------------------------------------------------------------
// Repack we_w2 (64 x 65536 f32) + we_b2 (65536 f32) into bf16 WMMA-B fragments:
//   fragment index = (c*8 + t)*32 + lane, 16 bf16 each (32 B)
//   lane l of tile t holds column w = t*16 + (l&15),
//   K rows kg = c*32 + 16*(l>>4) + e, e = 0..15
// ---------------------------------------------------------------------------
__global__ __launch_bounds__(256) void wtp_prep_kernel(
    const float* __restrict__ w2, const float* __restrict__ b2,
    unsigned short* __restrict__ wb)
{
    const int idx = blockIdx.x * 256 + threadIdx.x;      // one thread per fragment
    const int l  = idx & 31;
    const int t  = (idx >> 5) & 7;
    const int c  = idx >> 8;
    if (c >= NCHUNKS) return;
    const int w      = t * 16 + (l & 15);
    const int kgbase = c * 32 + 16 * (l >> 4);

    unsigned int pk[8];
#pragma unroll
    for (int e2 = 0; e2 < 8; ++e2) {
        unsigned int v = 0;
#pragma unroll
        for (int h = 0; h < 2; ++h) {
            const int kg = kgbase + e2 * 2 + h;
            float s;
            if (kg < 16384) {
                const int k = kg >> 8, uv = kg & 255;
                s = w2[(size_t)k * 65536 + uv * 128 + w];
            } else if (kg < 32768) {
                const int r = kg - 16384, k = r >> 8, uv = r & 255;
                s = w2[(size_t)k * 65536 + 32768 + uv * 128 + w];
            } else if (kg < 33024) {
                const int uv = kg - 32768;
                s = b2[uv * 128 + w];
            } else {
                const int uv = kg - 33024;
                s = b2[32768 + uv * 128 + w];
            }
            v |= (unsigned int)f2bf_rne(s) << (16 * h);
        }
        pk[e2] = v;
    }
    uint4* dst = (uint4*)(wb + (size_t)idx * 16);
    dst[0] = make_uint4(pk[0], pk[1], pk[2], pk[3]);
    dst[1] = make_uint4(pk[4], pk[5], pk[6], pk[7]);
}

// ---------------------------------------------------------------------------
// Branch-free GEMM inner loop over one type-uniform K segment.
//   kg_local = (c - segc)*32 ; k = kg_local>>8 ; uv0 = kg_local&255
//   A[m][K] = (USE_H ? hrow[k] : 1) * src[uv0 + Klocal]
// A fragment k order used here (k = 16h + s, same as B packing): lanes 0-15 hold K {0..15},
// lanes 16-31 hold K {16..31}  -> klo = 16*(lane>>4).
// ---------------------------------------------------------------------------
template <bool USE_H>
static __device__ __forceinline__ void gemm_segment(
    int ca, int cb, int segc,
    const float* __restrict__ src,     // ffp/ggp row base (16B aligned)
    const float* __restrict__ hrow,    // hbuf row base
    const unsigned short* __restrict__ wb,
    int nh, int lane, int klo,
    f32x8& a0, f32x8& a1, f32x8& a2, f32x8& a3)
{
    for (int c = ca; c < cb; ++c) {
        const int kg   = (c - segc) * 32;
        const int base = (kg & 255) + klo;            // multiple of 8 -> 16B aligned
        float hv = 1.f;
        if (USE_H) hv = hrow[kg >> 8];

        const float4 f0 = *(const float4*)(src + base);
        const float4 f1 = *(const float4*)(src + base + 4);
        const float4 f2 = *(const float4*)(src + base + 8);
        const float4 f3 = *(const float4*)(src + base + 12);

        bf16x16 a;
        a[0]  = (__bf16)(hv * f0.x); a[1]  = (__bf16)(hv * f0.y);
        a[2]  = (__bf16)(hv * f0.z); a[3]  = (__bf16)(hv * f0.w);
        a[4]  = (__bf16)(hv * f1.x); a[5]  = (__bf16)(hv * f1.y);
        a[6]  = (__bf16)(hv * f1.z); a[7]  = (__bf16)(hv * f1.w);
        a[8]  = (__bf16)(hv * f2.x); a[9]  = (__bf16)(hv * f2.y);
        a[10] = (__bf16)(hv * f2.z); a[11] = (__bf16)(hv * f2.w);
        a[12] = (__bf16)(hv * f3.x); a[13] = (__bf16)(hv * f3.y);
        a[14] = (__bf16)(hv * f3.z); a[15] = (__bf16)(hv * f3.w);

        const bf16x16* bp = (const bf16x16*)wb + (((size_t)c * 8 + nh * 4) * 32 + lane);
        __builtin_prefetch(bp + 8 * 32, 0, 0);        // next chunk's B fragments

        a0 = __builtin_amdgcn_wmma_f32_16x16x32_bf16(false, a, false, bp[0],      (short)0, a0, false, false);
        a1 = __builtin_amdgcn_wmma_f32_16x16x32_bf16(false, a, false, bp[32],     (short)0, a1, false, false);
        a2 = __builtin_amdgcn_wmma_f32_16x16x32_bf16(false, a, false, bp[64],     (short)0, a2, false, false);
        a3 = __builtin_amdgcn_wmma_f32_16x16x32_bf16(false, a, false, bp[96],     (short)0, a3, false, false);
    }
}

static __device__ __forceinline__ int imin(int a, int b) { return a < b ? a : b; }
static __device__ __forceinline__ int imax(int a, int b) { return a > b ? a : b; }

// ---------------------------------------------------------------------------
// Main fused kernel: one workgroup (256 threads = 8 waves, wave32) per 16 rows.
// ---------------------------------------------------------------------------
__global__ __launch_bounds__(256) void wtp_main_kernel(
    const float* __restrict__ x,
    const float* __restrict__ we_ln_g, const float* __restrict__ we_ln_b,
    const float* __restrict__ we_w1,
    const unsigned short* __restrict__ wb,
    const float* __restrict__ om_ln_g, const float* __restrict__ om_ln_b,
    const float* __restrict__ om_w1,  const float* __restrict__ om_w2,
    const float* __restrict__ om_b2,
    float* __restrict__ out)
{
    __shared__ float xbuf[16][65];          // raw x rows
    __shared__ float lnf [16][17];          // layernormed feats
    __shared__ float hbuf[16][65];          // silu hidden (16x64)
    __shared__ float ffp [16][FFSTRIDE];    // PATH_COEFF * f_u f_v
    __shared__ float ggp [16][FFSTRIDE];    // PATH_COEFF*CG * <x1_u, x1_v>
    __shared__ float tp  [16][130];         // tp_out (+2 cols for mu, rsig)
    __shared__ float outacc[16];

    const int tid  = threadIdx.x;
    const int row0 = blockIdx.x * BLK_ROWS;

    // ---- stage 1a: load x block, init accumulators ----
    for (int i = tid; i < 16 * 64; i += 256) {
        const int r = i >> 6, cidx = i & 63;
        xbuf[r][cidx] = x[(size_t)(row0 + r) * XD + cidx];
    }
    for (int i = tid; i < 16 * 128; i += 256) tp[i >> 7][i & 127] = 0.f;
    if (tid < 16) outacc[tid] = 0.f;
    __syncthreads();

    // ---- stage 1b: layernorm of feats (threads 0..15) ----
    if (tid < 16) {
        float mu = 0.f;
        for (int u = 0; u < 16; ++u) mu += xbuf[tid][u];
        mu *= (1.f / 16.f);
        float var = 0.f;
        for (int u = 0; u < 16; ++u) { const float d = xbuf[tid][u] - mu; var += d * d; }
        var *= (1.f / 16.f);
        const float rs = rsqrtf(var + 1e-5f);
        for (int u = 0; u < 16; ++u)
            lnf[tid][u] = (xbuf[tid][u] - mu) * rs * we_ln_g[u] + we_ln_b[u];
    }
    __syncthreads();

    // ---- stage 1c: h = silu(lnf @ we_w1), ffp, ggp ----
    for (int i = tid; i < 16 * 64; i += 256) {
        const int r = i >> 6, j = i & 63;
        float z = 0.f;
#pragma unroll
        for (int u = 0; u < 16; ++u) z += lnf[r][u] * we_w1[u * 64 + j];
        hbuf[r][j] = z / (1.f + __expf(-z));
    }
    for (int i = tid; i < 16 * 256; i += 256) {
        const int r = i >> 8, uv = i & 255, u = uv >> 4, v = uv & 15;
        ffp[r][uv] = PATH_COEFF * xbuf[r][u] * xbuf[r][v];
        const float* a = &xbuf[r][16 + u * 3];
        const float* b = &xbuf[r][16 + v * 3];
        ggp[r][uv] = (PATH_COEFF * CG110) * (a[0] * b[0] + a[1] * b[1] + a[2] * b[2]);
    }
    __syncthreads();

    // ---- stage 2: (16 x 33280) x (33280 x 128) bf16 WMMA GEMM ----
    // wave -> (K-quarter kq, N-half nh); 4 N-tiles per wave, 260 chunks per wave.
    const int wave = tid >> 5, lane = tid & 31;
    const int nh = wave >> 2, kq = wave & 3;
    const int m  = lane & 15;
    const int klo = 16 * (lane >> 4);

    f32x8 a0 = {}, a1 = {}, a2 = {}, a3 = {};

    const int c0 = kq * CHUNKS_PER_WAVE, c1 = c0 + CHUNKS_PER_WAVE;
    const float* ffr = &ffp[m][0];
    const float* ggr = &ggp[m][0];
    const float* hr  = &hbuf[m][0];

    // type-uniform segments: [0,512) ff*h | [512,1024) gg*h | [1024,1032) ff | [1032,1040) gg
    {   const int sa = c0,              sb = imin(c1, 512);
        if (sa < sb) gemm_segment<true >(sa, sb, 0,    ffr, hr, wb, nh, lane, klo, a0, a1, a2, a3); }
    {   const int sa = imax(c0, 512),   sb = imin(c1, 1024);
        if (sa < sb) gemm_segment<true >(sa, sb, 512,  ggr, hr, wb, nh, lane, klo, a0, a1, a2, a3); }
    {   const int sa = imax(c0, 1024),  sb = imin(c1, 1032);
        if (sa < sb) gemm_segment<false>(sa, sb, 1024, ffr, hr, wb, nh, lane, klo, a0, a1, a2, a3); }
    {   const int sa = imax(c0, 1032),  sb = c1;
        if (sa < sb) gemm_segment<false>(sa, sb, 1032, ggr, hr, wb, nh, lane, klo, a0, a1, a2, a3); }

    // reduce the 4 K-quarter partials into tp via LDS float atomics
    {
        const int rbase = (lane >> 4) * 8;     // C layout: vgpr r -> row r (+8 hi half)
        const int colb  = nh * 64 + (lane & 15);
#pragma unroll
        for (int r = 0; r < 8; ++r) {
            atomicAdd(&tp[rbase + r][colb +  0], a0[r]);
            atomicAdd(&tp[rbase + r][colb + 16], a1[r]);
            atomicAdd(&tp[rbase + r][colb + 32], a2[r]);
            atomicAdd(&tp[rbase + r][colb + 48], a3[r]);
        }
    }
    __syncthreads();

    // ---- stage 3: out = silu(LN(tp) @ om_w1) @ om_w2 + om_b2 ----
    if (tid < 16) {
        float s = 0.f, sq = 0.f;
        for (int u = 0; u < 128; ++u) { const float v = tp[tid][u]; s += v; sq += v * v; }
        const float mu = s * (1.f / 128.f);
        const float var = sq * (1.f / 128.f) - mu * mu;
        tp[tid][128] = mu;
        tp[tid][129] = rsqrtf(var + 1e-5f);
    }
    __syncthreads();
    for (int i = tid; i < 16 * 128; i += 256) {
        const int r = i >> 7, u = i & 127;
        tp[r][u] = (tp[r][u] - tp[r][128]) * tp[r][129] * om_ln_g[u] + om_ln_b[u];
    }
    __syncthreads();

    {
        const int r  = tid >> 4;              // 16 threads per row
        const int j0 = (tid & 15) * 32;       // 32 hidden units each
        float av = 0.f;
        for (int jj = 0; jj < 32; ++jj) {
            const int j = j0 + jj;
            float z = 0.f;
#pragma unroll 8
            for (int u = 0; u < 128; ++u) z += tp[r][u] * om_w1[u * 512 + j];
            av += (z / (1.f + __expf(-z))) * om_w2[j];
        }
        atomicAdd(&outacc[r], av);
    }
    __syncthreads();
    if (tid < 16) out[row0 + tid] = outacc[tid] + om_b2[0];
}

// ---------------------------------------------------------------------------
extern "C" void kernel_launch(void* const* d_in, const int* in_sizes, int n_in,
                              void* d_out, int out_size, void* d_ws, size_t ws_size,
                              hipStream_t stream) {
    (void)in_sizes; (void)n_in; (void)out_size; (void)ws_size;
    const float* x       = (const float*)d_in[0];
    const float* we_ln_g = (const float*)d_in[1];
    const float* we_ln_b = (const float*)d_in[2];
    const float* we_w1   = (const float*)d_in[3];
    const float* we_w2   = (const float*)d_in[4];
    const float* we_b2   = (const float*)d_in[5];
    const float* om_ln_g = (const float*)d_in[6];
    const float* om_ln_b = (const float*)d_in[7];
    const float* om_w1   = (const float*)d_in[8];
    const float* om_w2   = (const float*)d_in[9];
    const float* om_b2   = (const float*)d_in[10];
    float* out = (float*)d_out;

    unsigned short* wb = (unsigned short*)d_ws;   // 33280*128 bf16 = 8.5 MB

    wtp_prep_kernel<<<NCHUNKS, 256, 0, stream>>>(we_w2, we_b2, wb);
    wtp_main_kernel<<<NBLOCKS, 256, 0, stream>>>(
        x, we_ln_g, we_ln_b, we_w1, wb,
        om_ln_g, om_ln_b, om_w1, om_w2, om_b2, out);
}
